// DecoderWithAttention_11424613007900
// MI455X (gfx1250) — compile-verified, matrix operand order checked by reading, not run on hardware
//
#include <hip/hip_runtime.h>

// ---------------------------------------------------------------------------
// Problem constants (match reference)
// ---------------------------------------------------------------------------
constexpr int kB = 128, kP = 512, kD = 512, kH = 512, kA = 256, kE = 256,
              kV = 2000, kS = 160;
constexpr int kT  = kS - 1;    // 159 decode steps
constexpr int kX  = kE + kD;   // 768  (GRU1 input width)
constexpr int k3H = 3 * kH;    // 1536 (stacked r,z,n gates)

// ---------------------------------------------------------------------------
// CDNA5 WMMA types
// ---------------------------------------------------------------------------
typedef __attribute__((ext_vector_type(16))) __bf16       v16bf;
typedef __attribute__((ext_vector_type(8)))  float        v8f;
typedef __attribute__((ext_vector_type(8)))  unsigned int v8u;

__device__ __forceinline__ unsigned int pack_bf16(float lo, float hi) {
  // round-to-nearest-even f32 -> bf16, packed pair (lo in [15:0])
  unsigned int ul = __float_as_uint(lo), uh = __float_as_uint(hi);
  ul = (ul + 0x7FFFu + ((ul >> 16) & 1u)) >> 16;
  uh = (uh + 0x7FFFu + ((uh >> 16) & 1u)) >> 16;
  return (uh << 16) | (ul & 0xFFFFu);
}

// ---------------------------------------------------------------------------
// Batch sort by caption length (descending, stable) + decode lengths
// ---------------------------------------------------------------------------
__global__ void k_sort(const int* __restrict__ lens, int* __restrict__ sort_ind,
                       int* __restrict__ dl, float* __restrict__ out_dl,
                       float* __restrict__ out_si) {
  int i = threadIdx.x;  // one block, kB threads
  int li = lens[i];
  int rank = 0;
  for (int j = 0; j < kB; ++j) {
    int lj = lens[j];
    rank += (lj > li) || (lj == li && j < i);
  }
  sort_ind[rank] = i;
  __syncthreads();
  int s = sort_ind[i];
  int d = lens[s] - 1;
  dl[i]     = d;
  out_dl[i] = (float)d;
  out_si[i] = (float)s;
}

__global__ void k_caps_out(const int* __restrict__ caps,
                           const int* __restrict__ sort_ind,
                           float* __restrict__ out_caps) {
  int b = blockIdx.x;
  int src = sort_ind[b];
  for (int s = threadIdx.x; s < kS; s += blockDim.x)
    out_caps[b * kS + s] = (float)caps[src * kS + s];
}

// ---------------------------------------------------------------------------
// fp32 -> packed bf16 weight conversion (straight and transposing)
// ---------------------------------------------------------------------------
__global__ void k_cvt(const float* __restrict__ src, unsigned int* __restrict__ dst,
                      int npairs) {
  int i = blockIdx.x * blockDim.x + threadIdx.x;
  if (i < npairs) dst[i] = pack_bf16(src[2 * i], src[2 * i + 1]);
}

// src is [K,N] row-major; dst is [N][K/2] packed bf16 pairs (W^T, K-contiguous)
__global__ void k_cvt_t(const float* __restrict__ src, unsigned int* __restrict__ dst,
                        int N, int K) {
  int i = blockIdx.x * blockDim.x + threadIdx.x;
  int khalf = K >> 1;
  if (i >= N * khalf) return;
  int n = i / khalf, kp = i % khalf;
  dst[i] = pack_bf16(src[(2 * kp) * N + n], src[(2 * kp + 1) * N + n]);
}

// ---------------------------------------------------------------------------
// Fragment layouts per CDNA5 ISA 7.12.2 (v_wmma_f32_16x16x32_bf16):
//   A 16x32 bf16: lanes 0-15 rows M=0..15 hold K {0..7,16..23};
//                 lanes 16-31 same rows hold K {8..15,24..31}
//   C/D: VGPR v, lane l -> M = v + 8*(l>=16), N = l&15
// ---------------------------------------------------------------------------

// Hot-loop GEMM: C[M,N] = act( Abf[M,K](bf16) x W[N,K]^T(bf16) + bias[N] )
// One wave computes a 16x32 C macro-tile (two accumulators): the A fragment is
// loaded once per K-step and reused for two B tiles -> 3 loads per wmma.
// N must be a multiple of 16; the second N-tile is guarded (wave-uniform) so
// N=2000 works with grid.x = ceil(N/32).
__global__ void k_wmma_gemm_bf16a(const unsigned int* __restrict__ Abf, // [M][K/2]
                                  const unsigned int* __restrict__ W,   // [N][K/2]
                                  const float* __restrict__ bias,
                                  float* __restrict__ C, long ldc,
                                  int K, int N, int act,
                                  const int* __restrict__ dl, int t) {
  int lane = threadIdx.x;        // 32 lanes = one wave per block
  int hsel = lane >> 4;
  int l15  = lane & 15;
  int mbase = blockIdx.y * 16;
  int nbase = blockIdx.x * 32;
  int khalf = K >> 1;
  bool has2 = (nbase + 16) < N;                 // wave-uniform tail guard
  int n0 = nbase + l15;
  int n1 = has2 ? (nbase + 16 + l15) : n0;      // clamp loads in tail block
  const unsigned int* Arow = Abf + (long)(mbase + l15) * khalf;
  const unsigned int* W0   = W   + (long)n0 * khalf;
  const unsigned int* W1   = W   + (long)n1 * khalf;
  int off0 = hsel * 4, off1 = 8 + hsel * 4;

  v8f acc0 = {0.f, 0.f, 0.f, 0.f, 0.f, 0.f, 0.f, 0.f};
  v8f acc1 = {0.f, 0.f, 0.f, 0.f, 0.f, 0.f, 0.f, 0.f};
  for (int kp = 0; kp < khalf; kp += 16) {  // 32 K-elements per iteration
    const uint4 a0 = *(const uint4*)(Arow + kp + off0);
    const uint4 a1 = *(const uint4*)(Arow + kp + off1);
    const uint4 p0 = *(const uint4*)(W0 + kp + off0);
    const uint4 p1 = *(const uint4*)(W0 + kp + off1);
    const uint4 q0 = *(const uint4*)(W1 + kp + off0);
    const uint4 q1 = *(const uint4*)(W1 + kp + off1);
    v8u au, bu, cu;
    au[0] = a0.x; au[1] = a0.y; au[2] = a0.z; au[3] = a0.w;
    au[4] = a1.x; au[5] = a1.y; au[6] = a1.z; au[7] = a1.w;
    bu[0] = p0.x; bu[1] = p0.y; bu[2] = p0.z; bu[3] = p0.w;
    bu[4] = p1.x; bu[5] = p1.y; bu[6] = p1.z; bu[7] = p1.w;
    cu[0] = q0.x; cu[1] = q0.y; cu[2] = q0.z; cu[3] = q0.w;
    cu[4] = q1.x; cu[5] = q1.y; cu[6] = q1.z; cu[7] = q1.w;
    v16bf av = __builtin_bit_cast(v16bf, au);
    acc0 = __builtin_amdgcn_wmma_f32_16x16x32_bf16(
        false, av, false, __builtin_bit_cast(v16bf, bu),
        (short)0, acc0, false, false);
    acc1 = __builtin_amdgcn_wmma_f32_16x16x32_bf16(
        false, av, false, __builtin_bit_cast(v16bf, cu),
        (short)0, acc1, false, false);
  }

  int c0 = nbase + l15;
  float bv0 = bias[c0];
  float bv1 = has2 ? bias[c0 + 16] : 0.f;
#pragma unroll
  for (int v = 0; v < 8; ++v) {
    int r = mbase + v + hsel * 8;
    bool masked = dl && (t >= dl[r]);
    float v0 = acc0[v] + bv0;
    if (act == 1) v0 = fmaxf(v0, 0.f);
    if (masked) v0 = 0.f;
    C[(long)r * ldc + c0] = v0;
    if (has2) {
      float v1 = acc1[v] + bv1;
      if (act == 1) v1 = fmaxf(v1, 0.f);
      if (masked) v1 = 0.f;
      C[(long)r * ldc + c0 + 16] = v1;
    }
  }
}

// One-time GEMM with f32 A + row gather through sort_ind (encoder projection):
// att1[M,N] = A_gathered[M,K](f32->bf16) x W[N,K]^T(bf16) + bias[N]
__global__ void k_wmma_gemm_f32a(const float* __restrict__ Am, int lda,
                                 const unsigned int* __restrict__ W,  // [N][K/2]
                                 const float* __restrict__ bias,
                                 float* __restrict__ C, long ldc,
                                 int K,
                                 const int* __restrict__ sort_ind, int gatherP) {
  int lane = threadIdx.x;
  int hsel = lane >> 4;
  int l15  = lane & 15;
  int mbase = blockIdx.y * 16;
  int nbase = blockIdx.x * 16;

  int mrow = mbase + l15;
  long arow = (long)sort_ind[mrow / gatherP] * gatherP + (mrow % gatherP);
  const float* Arow = Am + arow * (long)lda;
  const unsigned int* Wrow = W + (long)(nbase + l15) * (K >> 1);
  int kh = hsel * 8;

  v8f acc = {0.f, 0.f, 0.f, 0.f, 0.f, 0.f, 0.f, 0.f};
  for (int k0 = 0; k0 < K; k0 += 32) {
    float4 a0 = *(const float4*)(Arow + k0 + kh);
    float4 a1 = *(const float4*)(Arow + k0 + kh + 4);
    float4 a2 = *(const float4*)(Arow + k0 + kh + 16);
    float4 a3 = *(const float4*)(Arow + k0 + kh + 20);
    v8u au;
    au[0] = pack_bf16(a0.x, a0.y); au[1] = pack_bf16(a0.z, a0.w);
    au[2] = pack_bf16(a1.x, a1.y); au[3] = pack_bf16(a1.z, a1.w);
    au[4] = pack_bf16(a2.x, a2.y); au[5] = pack_bf16(a2.z, a2.w);
    au[6] = pack_bf16(a3.x, a3.y); au[7] = pack_bf16(a3.z, a3.w);
    const uint4 b0 = *(const uint4*)(Wrow + (k0 >> 1) + hsel * 4);
    const uint4 b1 = *(const uint4*)(Wrow + (k0 >> 1) + 8 + hsel * 4);
    v8u bu;
    bu[0] = b0.x; bu[1] = b0.y; bu[2] = b0.z; bu[3] = b0.w;
    bu[4] = b1.x; bu[5] = b1.y; bu[6] = b1.z; bu[7] = b1.w;
    acc = __builtin_amdgcn_wmma_f32_16x16x32_bf16(
        false, __builtin_bit_cast(v16bf, au),
        false, __builtin_bit_cast(v16bf, bu),
        (short)0, acc, false, false);
  }

  int c = nbase + l15;
  float bv = bias[c];
#pragma unroll
  for (int v = 0; v < 8; ++v) {
    int r = mbase + v + hsel * 8;
    C[(long)r * ldc + c] = acc[v] + bv;
  }
}

// ---------------------------------------------------------------------------
// Attention: e = relu(att1 + att2) @ Wfa -> softmax -> context -> gate
// one block per (sorted) batch row; assembles x = [emb_t, gated_awe] in LDS
// and emits it as packed bf16 (GEMM A-operand format).
// ---------------------------------------------------------------------------
__global__ void k_attention(const float* __restrict__ att1,
                            const float* __restrict__ att2,
                            const float* __restrict__ g1,
                            const float* __restrict__ Wfa,
                            const float* __restrict__ bfa,
                            const float* __restrict__ Wg2,
                            const float* __restrict__ bg2,
                            const float* __restrict__ enc,
                            const int* __restrict__ sort_ind,
                            const int* __restrict__ caps,
                            const float* __restrict__ emb,
                            const int* __restrict__ dl, int t,
                            unsigned int* __restrict__ xbf,     // [B][kX/2]
                            float* __restrict__ out_alphas) {
  __shared__ float att2s[kA];
  __shared__ float sm[kP];
  __shared__ float rbuf[256];
  __shared__ float x_s[kX];
  __shared__ float gate_s, red_s;
  int b = blockIdx.x, tid = threadIdx.x;  // 256 threads

  att2s[tid] = att2[b * kA + tid];
  rbuf[tid]  = g1[b * kA + tid] * Wg2[tid];   // gate dot-product partials
  __syncthreads();
  for (int s = 128; s > 0; s >>= 1) {
    if (tid < s) rbuf[tid] += rbuf[tid + s];
    __syncthreads();
  }
  if (tid == 0) gate_s = 1.f / (1.f + expf(-(rbuf[0] + bg2[0])));

  // attention scores e[p]
  const float bf = bfa[0];
  for (int p = tid; p < kP; p += 256) {
    const float* a1 = att1 + ((long)b * kP + p) * kA;
    float accv = bf;
    for (int a = 0; a < kA; ++a) {
      float u = a1[a] + att2s[a];
      accv += fmaxf(u, 0.f) * Wfa[a];
    }
    sm[p] = accv;
  }
  __syncthreads();

  // softmax over P=512
  rbuf[tid] = fmaxf(sm[tid], sm[tid + 256]);
  __syncthreads();
  for (int s = 128; s > 0; s >>= 1) {
    if (tid < s) rbuf[tid] = fmaxf(rbuf[tid], rbuf[tid + s]);
    __syncthreads();
  }
  if (tid == 0) red_s = rbuf[0];
  __syncthreads();
  float mx = red_s;
  float e0 = expf(sm[tid] - mx), e1 = expf(sm[tid + 256] - mx);
  sm[tid] = e0; sm[tid + 256] = e1;
  rbuf[tid] = e0 + e1;
  __syncthreads();
  for (int s = 128; s > 0; s >>= 1) {
    if (tid < s) rbuf[tid] += rbuf[tid + s];
    __syncthreads();
  }
  if (tid == 0) red_s = 1.f / rbuf[0];
  __syncthreads();
  float inv = red_s;
  sm[tid] *= inv; sm[tid + 256] *= inv;
  __syncthreads();

  bool active = t < dl[b];
  out_alphas[((long)b * kT + t) * kP + tid]       = active ? sm[tid]       : 0.f;
  out_alphas[((long)b * kT + t) * kP + tid + 256] = active ? sm[tid + 256] : 0.f;

  // gated context vector -> x[E:E+D]
  float gate = gate_s;
  int srow = sort_ind[b];
  const float* encb = enc + (long)srow * kP * kD;
  for (int d = tid; d < kD; d += 256) {
    float s = 0.f;
    for (int p = 0; p < kP; ++p) s += sm[p] * encb[(long)p * kD + d];
    x_s[kE + d] = s * gate;
  }
  // teacher-forcing embedding -> x[0:E]
  int cap = caps[srow * kS + t];
  x_s[tid] = emb[(long)cap * kE + tid];
  __syncthreads();

  // emit packed bf16 x (A-operand for GRU1 / input-projection GEMMs)
  for (int i = tid; i < kX / 2; i += 256)
    xbf[b * (kX / 2) + i] = pack_bf16(x_s[2 * i], x_s[2 * i + 1]);
}

// ---------------------------------------------------------------------------
// GRU gate math + masked carry update + residual + LayerNorm.
// Emits: f32 carry h (in place), packed-bf16 carry copy (GEMM A for h@Whh^T),
//        f32 hr (residual chain), packed-bf16 hr (GEMM A for next layer).
// ---------------------------------------------------------------------------
__global__ void k_gru_ln(const float* __restrict__ gi, const float* __restrict__ gh,
                         float* __restrict__ h, unsigned int* __restrict__ hbf,
                         const float* __restrict__ resid,
                         const float* __restrict__ g, const float* __restrict__ beta,
                         const int* __restrict__ dl, int t,
                         float* __restrict__ hr, unsigned int* __restrict__ hrbf) {
  __shared__ float res_s[kH];
  __shared__ float h_s[kH];
  __shared__ float rbuf[256];
  __shared__ float mu_s, inv_s;
  int b = blockIdx.x, tid = threadIdx.x;
  bool active = t < dl[b];
  const float* gib = gi + (long)b * k3H;
  const float* ghb = gh + (long)b * k3H;
  for (int j = tid; j < kH; j += 256) {
    float r = 1.f / (1.f + expf(-(gib[j] + ghb[j])));
    float z = 1.f / (1.f + expf(-(gib[kH + j] + ghb[kH + j])));
    float n = tanhf(gib[2 * kH + j] + r * ghb[2 * kH + j]);
    float hp = h[b * kH + j];
    float hn = (1.f - z) * n + z * hp;
    float hc = active ? hn : hp;       // masked carry (ragged batch)
    if (active) h[b * kH + j] = hn;
    h_s[j] = hc;
    res_s[j] = hn + resid[b * kH + j]; // residual uses unmasked hn
  }
  __syncthreads();
  rbuf[tid] = res_s[tid] + res_s[tid + 256];
  __syncthreads();
  for (int s = 128; s > 0; s >>= 1) {
    if (tid < s) rbuf[tid] += rbuf[tid + s];
    __syncthreads();
  }
  if (tid == 0) mu_s = rbuf[0] * (1.f / kH);
  __syncthreads();
  float mu = mu_s;
  float d0 = res_s[tid] - mu, d1 = res_s[tid + 256] - mu;
  rbuf[tid] = d0 * d0 + d1 * d1;
  __syncthreads();
  for (int s = 128; s > 0; s >>= 1) {
    if (tid < s) rbuf[tid] += rbuf[tid + s];
    __syncthreads();
  }
  if (tid == 0) inv_s = rsqrtf(rbuf[0] * (1.f / kH) + 1e-5f);
  __syncthreads();
  float invv = inv_s;
  for (int j = tid; j < kH; j += 256) {
    float val = (res_s[j] - mu) * invv * g[j] + beta[j];
    hr[b * kH + j] = val;
    res_s[j] = val;                    // stage normalized value for packing
  }
  __syncthreads();
  // one bf16 pair per thread (kH/2 == 256)
  hrbf[b * (kH / 2) + tid] = pack_bf16(res_s[2 * tid], res_s[2 * tid + 1]);
  hbf [b * (kH / 2) + tid] = pack_bf16(h_s[2 * tid],  h_s[2 * tid + 1]);
}

// ---------------------------------------------------------------------------
// Host orchestration
// ---------------------------------------------------------------------------
extern "C" void kernel_launch(void* const* d_in, const int* in_sizes, int n_in,
                              void* d_out, int out_size, void* d_ws, size_t ws_size,
                              hipStream_t stream) {
  const float* enc  = (const float*)d_in[0];
  const int*   caps = (const int*)d_in[1];
  const int*   lens = (const int*)d_in[2];
  const float* Wea  = (const float*)d_in[3];  const float* bea  = (const float*)d_in[4];
  const float* Wda  = (const float*)d_in[5];  const float* bda  = (const float*)d_in[6];
  const float* Wfa  = (const float*)d_in[7];  const float* bfa  = (const float*)d_in[8];
  const float* Wg1  = (const float*)d_in[9];  const float* bg1  = (const float*)d_in[10];
  const float* Wg2  = (const float*)d_in[11]; const float* bg2  = (const float*)d_in[12];
  const float* emb  = (const float*)d_in[13];
  const float* Wih1 = (const float*)d_in[14]; const float* Whh1 = (const float*)d_in[15];
  const float* bih1 = (const float*)d_in[16]; const float* bhh1 = (const float*)d_in[17];
  const float* Wih2 = (const float*)d_in[18]; const float* Whh2 = (const float*)d_in[19];
  const float* bih2 = (const float*)d_in[20]; const float* bhh2 = (const float*)d_in[21];
  const float* Wih3 = (const float*)d_in[22]; const float* Whh3 = (const float*)d_in[23];
  const float* bih3 = (const float*)d_in[24]; const float* bhh3 = (const float*)d_in[25];
  const float* Wip  = (const float*)d_in[26]; const float* bip  = (const float*)d_in[27];
  const float* ln1g = (const float*)d_in[28]; const float* ln1b = (const float*)d_in[29];
  const float* ln2g = (const float*)d_in[30]; const float* ln2b = (const float*)d_in[31];
  const float* ln3g = (const float*)d_in[32]; const float* ln3b = (const float*)d_in[33];
  const float* Wfc  = (const float*)d_in[34]; const float* bfc  = (const float*)d_in[35];

  // tuple outputs, concatenated flat in return order
  float* out_preds  = (float*)d_out;
  float* out_caps   = out_preds  + (long)kB * kT * kV;
  float* out_dl     = out_caps   + (long)kB * kS;
  float* out_alphas = out_dl     + kB;
  float* out_si     = out_alphas + (long)kB * kT * kP;

  // ---- workspace bump allocator (~82 MB total) ----
  char* wp = (char*)d_ws;
  auto alloc = [&](size_t bytes) -> void* {
    void* r = wp;
    wp += (bytes + 255) & ~(size_t)255;
    return r;
  };
  int* si = (int*)alloc(kB * sizeof(int));
  int* dl = (int*)alloc(kB * sizeof(int));
  unsigned int* WeaT  = (unsigned int*)alloc(sizeof(int) * (size_t)kA * kD / 2);
  unsigned int* WdaT  = (unsigned int*)alloc(sizeof(int) * (size_t)kA * kH / 2);
  unsigned int* Wg1T  = (unsigned int*)alloc(sizeof(int) * (size_t)kA * kH / 2);
  unsigned int* WipT  = (unsigned int*)alloc(sizeof(int) * (size_t)kH * kX / 2);
  unsigned int* WfcT  = (unsigned int*)alloc(sizeof(int) * (size_t)kV * kH / 2);
  unsigned int* Wih1b = (unsigned int*)alloc(sizeof(int) * (size_t)k3H * kX / 2);
  unsigned int* Whh1b = (unsigned int*)alloc(sizeof(int) * (size_t)k3H * kH / 2);
  unsigned int* Wih2b = (unsigned int*)alloc(sizeof(int) * (size_t)k3H * kH / 2);
  unsigned int* Whh2b = (unsigned int*)alloc(sizeof(int) * (size_t)k3H * kH / 2);
  unsigned int* Wih3b = (unsigned int*)alloc(sizeof(int) * (size_t)k3H * kH / 2);
  unsigned int* Whh3b = (unsigned int*)alloc(sizeof(int) * (size_t)k3H * kH / 2);
  float* att1  = (float*)alloc(sizeof(float) * (size_t)kB * kP * kA);
  float* att2  = (float*)alloc(sizeof(float) * (size_t)kB * kA);
  float* g1buf = (float*)alloc(sizeof(float) * (size_t)kB * kA);
  float* xp    = (float*)alloc(sizeof(float) * (size_t)kB * kH);
  float* gi    = (float*)alloc(sizeof(float) * (size_t)kB * k3H);
  float* gh    = (float*)alloc(sizeof(float) * (size_t)kB * k3H);
  float* hbuf  = (float*)alloc(sizeof(float) * (size_t)3 * kB * kH);   // f32 carries
  float* h1 = hbuf, *h2 = hbuf + (size_t)kB * kH, *h3 = hbuf + (size_t)2 * kB * kH;
  float* h1r = (float*)alloc(sizeof(float) * (size_t)kB * kH);
  float* h2r = (float*)alloc(sizeof(float) * (size_t)kB * kH);
  float* h3r = (float*)alloc(sizeof(float) * (size_t)kB * kH);
  // packed-bf16 activation buffers (GEMM A operands)
  unsigned int* xbf  = (unsigned int*)alloc(sizeof(int) * (size_t)kB * kX / 2);
  unsigned int* hbcat = (unsigned int*)alloc(sizeof(int) * (size_t)3 * kB * kH / 2);
  unsigned int* hb1 = hbcat, *hb2 = hbcat + (size_t)kB * kH / 2,
               *hb3 = hbcat + (size_t)2 * kB * kH / 2;
  unsigned int* h1rb = (unsigned int*)alloc(sizeof(int) * (size_t)kB * kH / 2);
  unsigned int* h2rb = (unsigned int*)alloc(sizeof(int) * (size_t)kB * kH / 2);
  unsigned int* h3rb = (unsigned int*)alloc(sizeof(int) * (size_t)kB * kH / 2);

  auto cvt = [&](const float* src, unsigned int* dst, int npairs) {
    k_cvt<<<(npairs + 255) / 256, 256, 0, stream>>>(src, dst, npairs);
  };
  auto cvtt = [&](const float* src, unsigned int* dst, int N, int K) {
    int n = N * (K / 2);
    k_cvt_t<<<(n + 255) / 256, 256, 0, stream>>>(src, dst, N, K);
  };
  auto gemm = [&](const unsigned int* Abf, const unsigned int* W,
                  const float* bias, float* C, long ldc, int M, int N, int K,
                  int act, const int* dlp, int t) {
    dim3 g((N + 31) / 32, M / 16);
    k_wmma_gemm_bf16a<<<g, 32, 0, stream>>>(Abf, W, bias, C, ldc, K, N, act,
                                            dlp, t);
  };

  // ---- one-time setup ----
  k_sort<<<1, kB, 0, stream>>>(lens, si, dl, out_dl, out_si);
  k_caps_out<<<kB, 256, 0, stream>>>(caps, si, out_caps);

  cvtt(Wea, WeaT, kA, kD);     // [D,A] -> [A][D/2]
  cvtt(Wda, WdaT, kA, kH);
  cvtt(Wg1, Wg1T, kA, kH);
  cvtt(Wip, WipT, kH, kX);
  cvtt(Wfc, WfcT, kV, kH);
  cvt(Wih1, Wih1b, k3H * kX / 2);  // torch layout already [N,K]
  cvt(Whh1, Whh1b, k3H * kH / 2);
  cvt(Wih2, Wih2b, k3H * kH / 2);
  cvt(Whh2, Whh2b, k3H * kH / 2);
  cvt(Wih3, Wih3b, k3H * kH / 2);
  cvt(Whh3, Whh3b, k3H * kH / 2);

  // att1 = enc_sorted @ Wea + bea : one-time WMMA GEMM with row gather (f32 A)
  {
    dim3 g(kA / 16, (kB * kP) / 16);
    k_wmma_gemm_f32a<<<g, 32, 0, stream>>>(enc, kD, WeaT, bea, att1, kA, kD,
                                           si, kP);
  }

  // zero f32 + bf16 hidden-state carries (graph-capturable memset nodes)
  hipMemsetAsync(hbuf,  0, sizeof(float) * (size_t)3 * kB * kH, stream);
  hipMemsetAsync(hbcat, 0, sizeof(int) * (size_t)3 * kB * kH / 2, stream);

  // ---- recurrent decode loop ----
  for (int t = 0; t < kT; ++t) {
    gemm(hb2, WdaT, bda, att2, kA, kB, kA, kH, 0, nullptr, 0);
    gemm(hb2, Wg1T, bg1, g1buf, kA, kB, kA, kH, 1, nullptr, 0);
    k_attention<<<kB, 256, 0, stream>>>(att1, att2, g1buf, Wfa, bfa, Wg2, bg2,
                                        enc, si, caps, emb, dl, t, xbf,
                                        out_alphas);
    // layer 1
    gemm(xbf, Wih1b, bih1, gi, k3H, kB, k3H, kX, 0, nullptr, 0);
    gemm(hb1, Whh1b, bhh1, gh, k3H, kB, k3H, kH, 0, nullptr, 0);
    gemm(xbf, WipT,  bip,  xp, kH,  kB, kH,  kX, 0, nullptr, 0);
    k_gru_ln<<<kB, 256, 0, stream>>>(gi, gh, h1, hb1, xp, ln1g, ln1b, dl, t,
                                     h1r, h1rb);
    // layer 2
    gemm(h1rb, Wih2b, bih2, gi, k3H, kB, k3H, kH, 0, nullptr, 0);
    gemm(hb2,  Whh2b, bhh2, gh, k3H, kB, k3H, kH, 0, nullptr, 0);
    k_gru_ln<<<kB, 256, 0, stream>>>(gi, gh, h2, hb2, h1r, ln2g, ln2b, dl, t,
                                     h2r, h2rb);
    // layer 3
    gemm(h2rb, Wih3b, bih3, gi, k3H, kB, k3H, kH, 0, nullptr, 0);
    gemm(hb3,  Whh3b, bhh3, gh, k3H, kB, k3H, kH, 0, nullptr, 0);
    k_gru_ln<<<kB, 256, 0, stream>>>(gi, gh, h3, hb3, h2r, ln3g, ln3b, dl, t,
                                     h3r, h3rb);
    // logits with fused bias + decode-length mask, strided into [B,T,V]
    gemm(h3rb, WfcT, bfc, out_preds + (long)t * kV, (long)kT * kV,
         kB, kV, kH, 0, dl, t);
  }
}
